// MultiHeadAttention_85126251807436
// MI455X (gfx1250) — matrix operand order repaired and checked by reading, not compiled, not run on hardware
//
#include <hip/hip_runtime.h>
#include <hip/hip_bf16.h>

// ---------------------------------------------------------------------------
// CDNA5 (gfx1250) causal multi-head attention forward.
//   d_out = [ output (B*S*D f32) | attn_weights (B*H*S*S f32) ]
// f16 WMMA everywhere (v_wmma_f32_16x16x32_f16, f32 accumulate), 1/sqrt(HD)
// folded into Q, raw scores materialized into the attn_weights output buffer,
// causal softmax in place, then P*V and the output projection.
// Projection GEMMs use 128x128 block tiles (8 waves x 32x64 wave tile,
// 8 WMMA per k-step) with async global->LDS staging when available.
// ---------------------------------------------------------------------------

typedef __attribute__((ext_vector_type(16))) _Float16 v16h;
typedef __attribute__((ext_vector_type(8)))  _Float16 v8h;
typedef __attribute__((ext_vector_type(8)))  float    v8f;
typedef __attribute__((ext_vector_type(4)))  float    v4f;

#define WMMA_F16(a, b, c) \
  __builtin_amdgcn_wmma_f32_16x16x32_f16(false, (a), false, (b), (short)0, (c), false, false)

constexpr int B  = 2;
constexpr int S  = 2048;
constexpr int D  = 1024;
constexpr int H  = 16;
constexpr int HD = 64;      // head dim
constexpr int BH = B * H;   // 32
constexpr int MR = B * S;   // 4096 rows for the projection GEMMs
constexpr int LDSP = 40;    // padded LDS row stride in halves (80B, 16B aligned)

// gfx1250 async global->LDS path (ASYNCcnt-tracked, no VGPR round trip).
#if defined(__HIP_DEVICE_COMPILE__) && __has_builtin(__builtin_amdgcn_global_load_async_to_lds_b128)
#define USE_ASYNC_LDS 1
#endif

#if defined(USE_ASYNC_LDS)
// Builtin signature (from compiler diagnostic): param 1 is a pointer to
// int __attribute__((vector_size(16))) in the global (AS1) address space;
// param 2 is the LDS (AS3) pointer; then (imm offset, cpol).
typedef int v4i_g __attribute__((vector_size(16)));
typedef __attribute__((address_space(1))) v4i_g* as1_v4i_ptr;
typedef __attribute__((address_space(3))) v4i_g* as3_v4i_ptr;

__device__ __forceinline__ void async_copy16(const _Float16* g, _Float16* l) {
  // 8 halves = 128 bits. Generic->AS1 keeps the address; a generic LDS
  // pointer's low 32 bits are the LDS offset (flat aperture rule).
  __builtin_amdgcn_global_load_async_to_lds_b128(
      (as1_v4i_ptr)(size_t)g,
      (as3_v4i_ptr)(unsigned)(size_t)l,
      0, 0);
}
__device__ __forceinline__ void wait_async0() {
#if __has_builtin(__builtin_amdgcn_s_wait_asynccnt)
  __builtin_amdgcn_s_wait_asynccnt(0);
#else
  asm volatile("s_wait_asynccnt 0" ::: "memory");
#endif
}
#endif

// Combine two 8x f16 LDS/global loads into one 16x f16 WMMA fragment.
__device__ __forceinline__ v16h load2x8(const _Float16* p0, const _Float16* p1) {
  v8h lo = *(const v8h*)p0;
  v8h hi = *(const v8h*)p1;
  return __builtin_shufflevector(lo, hi, 0,1,2,3,4,5,6,7,8,9,10,11,12,13,14,15);
}

// ---------------------------------------------------------------------------
// 1) f32 -> f16 conversion, 8 elements per thread (n multiple of 8).
// ---------------------------------------------------------------------------
__global__ void cvt_f32_f16_kernel(const float* __restrict__ src,
                                   _Float16* __restrict__ dst, int n8) {
  int i = blockIdx.x * blockDim.x + threadIdx.x;
  if (i >= n8) return;
  const v4f* s = (const v4f*)src + 2 * (size_t)i;
  v4f a = s[0];
  v4f b = s[1];
  v8h h;
#pragma unroll
  for (int j = 0; j < 4; ++j) {
    h[j]     = (_Float16)a[j];
    h[j + 4] = (_Float16)b[j];
  }
  *((v8h*)dst + i) = h;
}

// ---------------------------------------------------------------------------
// 2) Generic linear: Y = X(MxK) * W(NxK)^T + bias, f16 in, f32 accumulate.
//    Block tile 128x128, 8 waves as 4(M) x 2(N); wave tile 32x64 ->
//    2 A-frags, 4 B-frags, 8 accumulators, 8 WMMA per 32-wide k-step.
//    PERHEAD=1: f16 per-head layout [B,H,S,HD], value*scale.
//    PERHEAD=0: f32 flat [M,N] (final output projection).
// ---------------------------------------------------------------------------
template <int PERHEAD>
__global__ void linear_wmma_kernel(const _Float16* __restrict__ X,
                                   const _Float16* __restrict__ Wt,
                                   const float* __restrict__ bias,
                                   float scale,
                                   float* __restrict__ outF,
                                   _Float16* __restrict__ outH) {
  constexpr int K = D;
  __shared__ __align__(16) _Float16 As[128 * LDSP];
  __shared__ __align__(16) _Float16 Bs[128 * LDSP];

  const int m0   = blockIdx.y * 128;
  const int n0   = blockIdx.x * 128;
  const int tid  = threadIdx.x;
  const int lane = tid & 31;
  const int wid  = tid >> 5;
  const int wm   = (wid >> 1) * 32;  // wave row offset in tile (0,32,64,96)
  const int wn   = (wid & 1) * 64;   // wave col offset in tile (0,64)
  const int lrow = lane & 15;
  const int lhalf = lane >> 4;

  const int srow = tid >> 1;         // staging: 128 rows, 2 threads/row
  const int scol = (tid & 1) * 16;   // each thread: 16 halves (2 x 16B)

  v8f acc[2][4] = {};

  for (int k0 = 0; k0 < K; k0 += 32) {
    __syncthreads();                 // previous tile fully consumed
    {
      const _Float16* ga = &X[(size_t)(m0 + srow) * K + k0 + scol];
      const _Float16* gb = &Wt[(size_t)(n0 + srow) * K + k0 + scol];
      _Float16* la = &As[srow * LDSP + scol];
      _Float16* lb = &Bs[srow * LDSP + scol];
#if defined(USE_ASYNC_LDS)
      async_copy16(ga,     la);
      async_copy16(ga + 8, la + 8);
      async_copy16(gb,     lb);
      async_copy16(gb + 8, lb + 8);
      wait_async0();
#else
      *(v8h*)la       = *(const v8h*)ga;
      *(v8h*)(la + 8) = *(const v8h*)(ga + 8);
      *(v8h*)lb       = *(const v8h*)gb;
      *(v8h*)(lb + 8) = *(const v8h*)(gb + 8);
#endif
    }
    __syncthreads();                 // tile visible to all waves

    v16h a[2];
#pragma unroll
    for (int i = 0; i < 2; ++i) {
      const _Float16* abase = &As[(wm + i * 16 + lrow) * LDSP];
      a[i] = load2x8(abase + lhalf * 16, abase + lhalf * 16 + 8);
    }
    v16h bf[4];
#pragma unroll
    for (int j = 0; j < 4; ++j) {
      const _Float16* bp = &Bs[(wn + j * 16 + lrow) * LDSP + lhalf * 16];
      bf[j] = load2x8(bp, bp + 8);
    }
#pragma unroll
    for (int i = 0; i < 2; ++i)
#pragma unroll
      for (int j = 0; j < 4; ++j)
        acc[i][j] = WMMA_F16(a[i], bf[j], acc[i][j]);
  }

#pragma unroll
  for (int i = 0; i < 2; ++i) {
#pragma unroll
    for (int j = 0; j < 4; ++j) {
#pragma unroll
      for (int v = 0; v < 8; ++v) {
        const int m = m0 + wm + i * 16 + v + 8 * lhalf;
        const int n = n0 + wn + j * 16 + lrow;
        const float r = (acc[i][j][v] + bias[n]) * scale;
        if (PERHEAD) {
          const int bb = m / S, ss = m % S;
          outH[((size_t)(bb * H + n / HD) * S + ss) * HD + (n % HD)] = (_Float16)r;
        } else {
          outF[(size_t)m * D + n] = r;
        }
      }
    }
  }
}

// ---------------------------------------------------------------------------
// 3) Scores: raw S_scores[bh, q, k] = Q[bh,q,:] . K[bh,k,:] (scale in Q).
//    Upper-triangular 64x64 tiles skipped (softmax zeros them). Q/K rows are
//    contiguous 128B -> WMMA fragments straight from global, no LDS.
// ---------------------------------------------------------------------------
__global__ void attn_scores_kernel(const _Float16* __restrict__ Qh,
                                   const _Float16* __restrict__ Kh,
                                   float* __restrict__ scores) {
  const int kt = blockIdx.x;
  const int qt = blockIdx.y;
  const int bh = blockIdx.z;
  if (kt > qt) return;                 // strictly above the diagonal

  const int q0 = qt * 64;
  const int k0 = kt * 64;
  const int tid  = threadIdx.x;
  const int lane = tid & 31;
  const int wid  = tid >> 5;
  const int wm   = (wid >> 1) * 16;
  const int wn   = (wid & 1) * 32;
  const int lrow = lane & 15;
  const int lhalf = lane >> 4;

  const _Float16* Qb = Qh + (size_t)bh * S * HD;
  const _Float16* Kb = Kh + (size_t)bh * S * HD;

  v8f acc0 = {};
  v8f acc1 = {};

#pragma unroll
  for (int kk = 0; kk < HD; kk += 32) {
    const _Float16* abase = Qb + (size_t)(q0 + wm + lrow) * HD + kk;
    v16h a = load2x8(abase + lhalf * 16, abase + lhalf * 16 + 8);

    const _Float16* b0p = Kb + (size_t)(k0 + wn + lrow) * HD + kk + lhalf * 16;
    v16h b0 = load2x8(b0p, b0p + 8);
    const _Float16* b1p = Kb + (size_t)(k0 + wn + 16 + lrow) * HD + kk + lhalf * 16;
    v16h b1 = load2x8(b1p, b1p + 8);

    acc0 = WMMA_F16(a, b0, acc0);
    acc1 = WMMA_F16(a, b1, acc1);
  }

#pragma unroll
  for (int v = 0; v < 8; ++v) {
    const int q = q0 + wm + v + 8 * lhalf;
    float* row = scores + ((size_t)bh * S + q) * S;
    row[k0 + wn + lrow]      = acc0[v];
    row[k0 + wn + 16 + lrow] = acc1[v];
  }
}

// ---------------------------------------------------------------------------
// 4) Causal softmax, in place, one 256-thread block per row.
//    Reduces over j <= q only; writes zeros for j > q.
// ---------------------------------------------------------------------------
__global__ void softmax_causal_kernel(float* __restrict__ att) {
  const size_t row = blockIdx.x;              // 0 .. BH*S-1
  const int q = (int)(row % S);
  float* ptr = att + row * (size_t)S;
  const int n = q + 1;
  const int t = threadIdx.x;

  __shared__ float red[256];

  float m = -3.0e38f;
  for (int j = t; j < n; j += 256) m = fmaxf(m, ptr[j]);
  red[t] = m;
  __syncthreads();
  for (int s2 = 128; s2 > 0; s2 >>= 1) {
    if (t < s2) red[t] = fmaxf(red[t], red[t + s2]);
    __syncthreads();
  }
  m = red[0];
  __syncthreads();

  float sum = 0.0f;
  for (int j = t; j < n; j += 256) {
    const float e = __expf(ptr[j] - m);
    ptr[j] = e;
    sum += e;
  }
  red[t] = sum;
  __syncthreads();
  for (int s2 = 128; s2 > 0; s2 >>= 1) {
    if (t < s2) red[t] += red[t + s2];
    __syncthreads();
  }
  const float inv = 1.0f / red[0];
  __syncthreads();

  for (int j = t; j < n; j += 256) ptr[j] *= inv;
  for (int j = n + t; j < S; j += 256) ptr[j] = 0.0f;  // masked region
}

// ---------------------------------------------------------------------------
// 5) PV: attn_out[b, q, h*HD+hd] = sum_k P[bh,q,k] * V[bh,k,hd]  (f16 out).
//    P staged f32->f16 into LDS; V staged transposed ([hd][k]) so the WMMA B
//    fragment reads K-contiguous. Causal bound: k < q0+64 (P is zero beyond q).
// ---------------------------------------------------------------------------
__global__ void attn_pv_kernel(const float* __restrict__ P,
                               const _Float16* __restrict__ Vh,
                               _Float16* __restrict__ AO) {
  const int qt = blockIdx.x;
  const int bh = blockIdx.y;
  const int q0 = qt * 64;

  __shared__ __align__(16) _Float16 Ps[64 * LDSP];  // [q][k]
  __shared__ __align__(16) _Float16 Vs[64 * LDSP];  // [hd][k] (transposed)

  const int tid  = threadIdx.x;
  const int lane = tid & 31;
  const int wid  = tid >> 5;
  const int wm   = (wid >> 1) * 16;   // q offset
  const int wn   = (wid & 1) * 32;    // hd offset
  const int lrow = lane & 15;
  const int lhalf = lane >> 4;

  const int srow = tid >> 2;          // P staging: 64 q-rows
  const int scol = (tid & 3) * 8;     // 8 f32 per thread
  const int vk   = tid & 31;          // V staging: key within k-tile
  const int vh0  = (tid >> 5) * 8;    // 8 hd per thread

  const float*    Pb = P  + ((size_t)bh * S + q0) * S;
  const _Float16* Vb = Vh + (size_t)bh * S * HD;

  v8f acc0 = {};
  v8f acc1 = {};

  const int kmax = q0 + 64;           // causal: keys beyond q0+63 contribute 0
  for (int k0 = 0; k0 < kmax; k0 += 32) {
    __syncthreads();
    {   // P tile 64x32, f32 -> f16
      const float* src = Pb + (size_t)srow * S + k0 + scol;
      v4f p0 = *(const v4f*)src;
      v4f p1 = *(const v4f*)(src + 4);
      v8h h;
#pragma unroll
      for (int j = 0; j < 4; ++j) {
        h[j]     = (_Float16)p0[j];
        h[j + 4] = (_Float16)p1[j];
      }
      *(v8h*)&Ps[srow * LDSP + scol] = h;
    }
    {   // V tile 32x64 -> transposed LDS [hd][k]
      v8h vv = *(const v8h*)&Vb[(size_t)(k0 + vk) * HD + vh0];
#pragma unroll
      for (int j = 0; j < 8; ++j) Vs[(vh0 + j) * LDSP + vk] = vv[j];
    }
    __syncthreads();

    const _Float16* abase = &Ps[(wm + lrow) * LDSP];
    v16h a = load2x8(abase + lhalf * 16, abase + lhalf * 16 + 8);

    const _Float16* b0p = &Vs[(wn + lrow) * LDSP + lhalf * 16];
    v16h b0 = load2x8(b0p, b0p + 8);
    const _Float16* b1p = &Vs[(wn + 16 + lrow) * LDSP + lhalf * 16];
    v16h b1 = load2x8(b1p, b1p + 8);

    acc0 = WMMA_F16(a, b0, acc0);
    acc1 = WMMA_F16(a, b1, acc1);
  }

  const int bb = bh / H;
  const int hh = bh % H;
#pragma unroll
  for (int v = 0; v < 8; ++v) {
    const int q = q0 + wm + v + 8 * lhalf;
    const size_t rowoff = ((size_t)bb * S + q) * D + hh * HD;
    AO[rowoff + wn + lrow]      = (_Float16)acc0[v];
    AO[rowoff + wn + 16 + lrow] = (_Float16)acc1[v];
  }
}

// ---------------------------------------------------------------------------
// Launch
// ---------------------------------------------------------------------------
extern "C" void kernel_launch(void* const* d_in, const int* in_sizes, int n_in,
                              void* d_out, int out_size, void* d_ws, size_t ws_size,
                              hipStream_t stream) {
  const float* x  = (const float*)d_in[0];
  const float* Wq = (const float*)d_in[1];
  const float* bq = (const float*)d_in[2];
  const float* Wk = (const float*)d_in[3];
  const float* bk = (const float*)d_in[4];
  const float* Wv = (const float*)d_in[5];
  const float* bv = (const float*)d_in[6];
  const float* Wo = (const float*)d_in[7];
  const float* bo = (const float*)d_in[8];

  float* out  = (float*)d_out;                   // [B,S,D]
  float* attw = out + (size_t)B * S * D;         // [B,H,S,S]

  // Carve f16 scratch from d_ws (256B aligned chunks).
  char*  base = (char*)d_ws;
  size_t off  = 0;
  auto take = [&](size_t bytes) -> void* {
    void* p = base + off;
    off += (bytes + 255) & ~(size_t)255;
    return p;
  };
  _Float16* xh  = (_Float16*)take((size_t)B * S * D * 2);
  _Float16* Wqh = (_Float16*)take((size_t)D * D * 2);
  _Float16* Wkh = (_Float16*)take((size_t)D * D * 2);
  _Float16* Wvh = (_Float16*)take((size_t)D * D * 2);
  _Float16* Woh = (_Float16*)take((size_t)D * D * 2);
  _Float16* Qh  = (_Float16*)take((size_t)B * S * D * 2);   // [B,H,S,HD]
  _Float16* Kh  = (_Float16*)take((size_t)B * S * D * 2);
  _Float16* Vh  = (_Float16*)take((size_t)B * S * D * 2);
  _Float16* AOh = (_Float16*)take((size_t)B * S * D * 2);   // merged-head attn out

  // 1) f32 -> f16 conversions
  {
    const int nx = B * S * D / 8;   // 524288
    const int nw = D * D / 8;       // 131072
    cvt_f32_f16_kernel<<<(nx + 255) / 256, 256, 0, stream>>>(x,  xh,  nx);
    cvt_f32_f16_kernel<<<(nw + 255) / 256, 256, 0, stream>>>(Wq, Wqh, nw);
    cvt_f32_f16_kernel<<<(nw + 255) / 256, 256, 0, stream>>>(Wk, Wkh, nw);
    cvt_f32_f16_kernel<<<(nw + 255) / 256, 256, 0, stream>>>(Wv, Wvh, nw);
    cvt_f32_f16_kernel<<<(nw + 255) / 256, 256, 0, stream>>>(Wo, Woh, nw);
  }

  // 2) Q/K/V projections (per-head f16). 1/sqrt(HD) folded into Q.
  {
    dim3 grid(D / 128, MR / 128);   // 8 x 32 = 256 blocks (~1 per WGP)
    const float qscale = 0.125f;    // 1/sqrt(64)
    linear_wmma_kernel<1><<<grid, 256, 0, stream>>>(xh, Wqh, bq, qscale, nullptr, Qh);
    linear_wmma_kernel<1><<<grid, 256, 0, stream>>>(xh, Wkh, bk, 1.0f,   nullptr, Kh);
    linear_wmma_kernel<1><<<grid, 256, 0, stream>>>(xh, Wvh, bv, 1.0f,   nullptr, Vh);
  }

  // 3) Raw scores into attn_weights buffer (upper-tri tiles skipped).
  attn_scores_kernel<<<dim3(S / 64, S / 64, BH), 256, 0, stream>>>(Qh, Kh, attw);

  // 4) Causal softmax in place (also zeros the masked region).
  softmax_causal_kernel<<<dim3(BH * S), 256, 0, stream>>>(attw);

  // 5) P @ V -> merged-head f16 attn output.
  attn_pv_kernel<<<dim3(S / 64, BH), 256, 0, stream>>>(attw, Vh, AOh);

  // 6) Output projection -> f32 d_out.
  linear_wmma_kernel<0><<<dim3(D / 128, MR / 128), 256, 0, stream>>>(AOh, Woh, bo, 1.0f, out, nullptr);
}
